// Net_34900904247300
// MI455X (gfx1250) — matrix operand order repaired and checked by reading, not compiled, not run on hardware
//
#include <hip/hip_runtime.h>

// MI455X (gfx1250): wave32, WMMA bf16 16x16x32 for the similarity GEMM.
// Fused 3-iteration clustering, with CDNA5 async global->LDS double-buffered
// staging of the codebook tiles (ASYNCcnt pipeline).

typedef __attribute__((ext_vector_type(16))) __bf16 v16bf;
typedef __attribute__((ext_vector_type(8)))  float  v8f;

#define D_DIM   512
#define V_DIM   1000
#define VP_DIM  1008          // padded to 63 N-tiles of 16
#define VT      63
#define T_DIM   2048
#define B_DIM   16
#define NIT     3
#define KC      16            // 512 / 32 K-chunks

__device__ __forceinline__ unsigned short f2bf(float f) {
    unsigned int u = __float_as_uint(f);
    u += 0x7FFFu + ((u >> 16) & 1u);          // round-to-nearest-even
    return (unsigned short)(u >> 16);
}

// ---------------------------------------------------------------------------
// Prep: Wb[v][d] = bf16( W[v][d] / ||W_v|| )   (norm folded into the operand,
// so the GEMM result is directly the argmax score). Rows >= 1000 zero-padded.
// ---------------------------------------------------------------------------
__global__ __launch_bounds__(128) void prep_wb(const float* __restrict__ W,
                                               unsigned short* __restrict__ Wb) {
    const int v = blockIdx.x;                  // 0..1007
    __shared__ float red[4];
    float s = 0.0f;
    if (v < V_DIM) {
        const float* wr = W + (size_t)v * D_DIM;
        for (int d = threadIdx.x; d < D_DIM; d += 128) { float w = wr[d]; s += w * w; }
    }
    #pragma unroll
    for (int off = 16; off >= 1; off >>= 1) s += __shfl_xor(s, off, 32);
    if ((threadIdx.x & 31) == 0) red[threadIdx.x >> 5] = s;
    __syncthreads();
    s = red[0] + red[1] + red[2] + red[3];
    unsigned short* br = Wb + (size_t)v * D_DIM;
    if (v < V_DIM) {
        const float inv = rsqrtf(s);
        const float* wr = W + (size_t)v * D_DIM;
        for (int d = threadIdx.x; d < D_DIM; d += 128) br[d] = f2bf(wr[d] * inv);
    } else {
        for (int d = threadIdx.x; d < D_DIM; d += 128) br[d] = 0;
    }
}

// ---------------------------------------------------------------------------
// Async DMA of one 16x512 bf16 codebook tile (16 KB) straight into LDS.
// 256 threads x 4 x B128 => 4 async-load instructions per wave (ASYNCcnt += 4).
// ---------------------------------------------------------------------------
__device__ __forceinline__ void issue_tile_async(const unsigned short* __restrict__ Wb,
                                                 int j, unsigned lds_base, int tid) {
    const uint4* src = (const uint4*)(Wb + (size_t)j * 16 * D_DIM);
    #pragma unroll
    for (int i = 0; i < 4; ++i) {
        const uint4* gp = src + (tid + 256 * i);
        unsigned     lo = lds_base + (unsigned)(tid + 256 * i) * 16u;
        asm volatile("global_load_async_to_lds_b128 %0, %1, off"
                     :: "v"(lo), "v"(gp) : "memory");
    }
}

// ---------------------------------------------------------------------------
// Fused kernel: 256 threads = 8 waves, 16 rows per wave, 128 rows per block.
// ---------------------------------------------------------------------------
__global__ __launch_bounds__(256) void cluster_fused(
    const float* __restrict__ W,               // [1000][512] fp32 (for gather)
    const unsigned short* __restrict__ Wb,     // [1008][512] bf16 (normalized)
    float* __restrict__ xcur,                  // [32768][512] fp32 residual
    float* __restrict__ out)                   // [16][3][2048][512] fp32
{
    __shared__ unsigned short bs[2][16 * D_DIM];   // 2 x 16 KB double buffer

    const int tid   = threadIdx.x;
    const int lane  = tid & 31;
    const int wave  = tid >> 5;
    const int lo    = lane & 15;
    const int hi    = lane >> 4;
    const int wbase = blockIdx.x * 128 + wave * 16;   // this wave's 16 rows

    unsigned lbase[2];
    lbase[0] = (unsigned)(size_t)&bs[0][0];    // flat->LDS: low 32 bits = LDS addr
    lbase[1] = (unsigned)(size_t)&bs[1][0];

    // prime the pipeline: tile 0 of iteration 0 into buffer 0
    issue_tile_async(Wb, 0, lbase[0], tid);
    int buf = 0;

    for (int it = 0; it < NIT; ++it) {
        // ---- A fragments: 16 rows x 512 K, bf16, full K resident in regs ----
        // (overlaps with the in-flight async tile load)
        v16bf afrag[KC];
        {
            const float* xr = xcur + (size_t)(wbase + lo) * D_DIM;
            #pragma unroll
            for (int c = 0; c < KC; ++c) {
                union { v16bf v; unsigned short s[16]; } ua;
                const int k0 = 32 * c + hi * 16;       // VGPR0..3 K-run (k = 16h + s)
                const int k1 = 32 * c + hi * 16 + 8;   // VGPR4..7 K-run (k = 16h + s)
                #pragma unroll
                for (int i = 0; i < 8; ++i) ua.s[i]     = f2bf(xr[k0 + i]);
                #pragma unroll
                for (int i = 0; i < 8; ++i) ua.s[8 + i] = f2bf(xr[k1 + i]);
                afrag[c] = ua.v;
            }
        }

        float best[8];
        int   bidx[8];
        #pragma unroll
        for (int r = 0; r < 8; ++r) { best[r] = -3.0e38f; bidx[r] = 0; }

        for (int j = 0; j < VT; ++j) {
            // keep one tile in flight: issue j+1 (wrapping to next iteration's
            // tile 0, which then overlaps the whole epilogue below)
            const bool more = !(it == NIT - 1 && j == VT - 1);
            if (more) {
                const int nj = (j + 1 == VT) ? 0 : j + 1;
                issue_tile_async(Wb, nj, lbase[buf ^ 1], tid);
                asm volatile("s_wait_asynccnt 4" ::: "memory");  // tile j landed
            } else {
                asm volatile("s_wait_asynccnt 0" ::: "memory");
            }
            __syncthreads();                       // tile j visible to all waves

            v8f acc = {};
            const unsigned short* bp = &bs[buf][0] + lo * D_DIM + hi * 16;
            #pragma unroll
            for (int c = 0; c < KC; ++c) {
                union { v16bf v; uint4 q[2]; } ub;
                const uint4* q = (const uint4*)(bp + c * 32);     // 32 contiguous B
                ub.q[0] = q[0];
                ub.q[1] = q[1];
                acc = __builtin_amdgcn_wmma_f32_16x16x32_bf16(
                        false, afrag[c], false, ub.v, (short)0, acc, false, false);
            }

            const int  v     = j * 16 + lo;        // N index held by this lane
            const bool valid = v < V_DIM;
            #pragma unroll
            for (int r = 0; r < 8; ++r) {          // C vgpr r -> row r + hi*8
                const float cv = acc[r];
                if (valid && (cv > best[r])) { best[r] = cv; bidx[r] = v; }
            }
            __syncthreads();                       // buffer free for tile j+2
            buf ^= 1;
        }

        // ---- argmax across the 16 lanes of each half-wave (tie -> low idx) --
        #pragma unroll
        for (int r = 0; r < 8; ++r) {
            #pragma unroll
            for (int off = 8; off >= 1; off >>= 1) {
                const float ov = __shfl_xor(best[r], off, 16);
                const int   oi = __shfl_xor(bidx[r], off, 16);
                if (ov > best[r] || (ov == best[r] && oi < bidx[r])) {
                    best[r] = ov; bidx[r] = oi;
                }
            }
        }
        int rowidx[16];                            // broadcast to whole wave
        #pragma unroll
        for (int m = 0; m < 16; ++m)
            rowidx[m] = __shfl(bidx[m & 7], (m >> 3) * 16, 32);

        // ---- epilogue: a = W[idx]; softmax(a*x) * a; x -= a; write anchor ---
        // (the next iteration's tile 0 async load is in flight during this)
        const bool last = (it == NIT - 1);
        for (int m = 0; m < 16; ++m) {
            const int grow = wbase + m;
            const float* wr = W + (size_t)rowidx[m] * D_DIM;
            float* xr = xcur + (size_t)grow * D_DIM;

            float av[16], xv[16], tv[16];
            float mx = -3.0e38f;
            #pragma unroll
            for (int i = 0; i < 16; ++i) {
                const int d = lane * 16 + i;
                av[i] = wr[d];
                xv[i] = xr[d];
                tv[i] = av[i] * xv[i];
                mx = fmaxf(mx, tv[i]);
            }
            #pragma unroll
            for (int off = 16; off >= 1; off >>= 1)
                mx = fmaxf(mx, __shfl_xor(mx, off, 32));
            float sum = 0.0f;
            #pragma unroll
            for (int i = 0; i < 16; ++i) { tv[i] = __expf(tv[i] - mx); sum += tv[i]; }
            #pragma unroll
            for (int off = 16; off >= 1; off >>= 1)
                sum += __shfl_xor(sum, off, 32);
            const float inv = 1.0f / sum;

            const int b  = grow >> 11;             // row / 2048
            const int tt = grow & (T_DIM - 1);
            float* orow = out + (((size_t)b * NIT + it) * T_DIM + tt) * (size_t)D_DIM;
            #pragma unroll
            for (int i = 0; i < 16; ++i) {
                const int d = lane * 16 + i;
                const float anch = tv[i] * inv * av[i];
                orow[d] = anch;
                if (!last) xr[d] = xv[i] - anch;   // residual for next iteration
            }
        }
    }
}

extern "C" void kernel_launch(void* const* d_in, const int* in_sizes, int n_in,
                              void* d_out, int out_size, void* d_ws, size_t ws_size,
                              hipStream_t stream) {
    const float* x = (const float*)d_in[0];        // [16,2048,512] fp32
    const float* W = (const float*)d_in[1];        // [1000,512]    fp32
    float* out = (float*)d_out;                    // [16,3,2048,512] fp32

    // workspace layout: [0,1MB) normalized bf16 codebook, [1MB, 1MB+64MB) residual x
    unsigned short* Wb = (unsigned short*)d_ws;
    float* xcur = (float*)((char*)d_ws + (size_t)(1 << 20));

    hipMemcpyAsync(xcur, x, (size_t)B_DIM * T_DIM * D_DIM * sizeof(float),
                   hipMemcpyDeviceToDevice, stream);
    prep_wb<<<VP_DIM, 128, 0, stream>>>(W, Wb);
    cluster_fused<<<(B_DIM * T_DIM) / 128, 256, 0, stream>>>(W, Wb, xcur, out);
}
